// SelfAttention_65249143161555
// MI455X (gfx1250) — matrix operand order repaired and checked by reading, not compiled, not run on hardware
//
#include <hip/hip_runtime.h>
#include <hip/hip_bf16.h>
#include <stdint.h>

// ---------------------------------------------------------------------------
// Self-attention, B=8 S=2048 D=256, fp32 in/out.
// bf16 WMMA (v_wmma_f32_16x16x32_bf16) for all matmuls, fp32 flash-softmax.
// Kernel 1 projects E->Q,K (row-major bf16) and V->VT (transposed bf16).
// Kernel 2: causal flash attention; K-tiles staged once per block into LDS
// with double-buffered async global->LDS copies (ASYNCcnt) when available.
// ---------------------------------------------------------------------------

typedef __attribute__((ext_vector_type(16))) __bf16 v16bf;
typedef __attribute__((ext_vector_type(8)))  float  v8f;
typedef __attribute__((ext_vector_type(4)))  int    v4i;

#if defined(__has_builtin)
#  if __has_builtin(__builtin_amdgcn_global_load_async_to_lds_b128)
#    define HAVE_ASYNC_LDS 1
#  endif
#  if __has_builtin(__builtin_amdgcn_s_wait_asynccnt)
#    define HAVE_WAIT_ASYNC 1
#  endif
#endif
#ifndef HAVE_ASYNC_LDS
#  define HAVE_ASYNC_LDS 0
#endif
#ifndef HAVE_WAIT_ASYNC
#  define HAVE_WAIT_ASYNC 0
#endif

typedef __attribute__((address_space(1))) v4i* gas_v4i;   // global (AS1)
typedef __attribute__((address_space(3))) v4i* las_v4i;   // LDS    (AS3)

union Frag {
  unsigned int u[8];
  uint4        q2[2];
  v16bf        v;
};

__device__ __forceinline__ unsigned short f2bf(float f) {
  unsigned int u = __float_as_uint(f);
  u += 0x7FFFu + ((u >> 16) & 1u);     // round-to-nearest-even
  return (unsigned short)(u >> 16);
}

__device__ __forceinline__ v8f wmma_bf16(const Frag& a, const Frag& b, v8f c) {
  return __builtin_amdgcn_wmma_f32_16x16x32_bf16(
      /*neg_a=*/false, a.v, /*neg_b=*/false, b.v,
      /*c_mod=*/(short)0, c, /*reuse_a=*/false, /*reuse_b=*/false);
}

__device__ __forceinline__ void wait_async_all() {
#if HAVE_ASYNC_LDS
#  if HAVE_WAIT_ASYNC
  __builtin_amdgcn_s_wait_asynccnt(0);
#  else
  asm volatile("s_wait_asynccnt 0x0" ::: "memory");
#  endif
#endif
}

#define SEQ   2048
#define DIM   256
#define NROWS 16384   // B*S

// ---------------------------------------------------------------------------
// Kernel 1: Q = E*WQ^T, K = E*WK^T, V = E*WV^T  (C[m,n] = sum_k X[m,k]W[n,k])
// grid = (NROWS/64, DIM/64, 3), block = 256 (8 waves). Block tile 64x64.
// z==2 (V): store transposed as VT[b][d][s] bf16.
// ---------------------------------------------------------------------------
__global__ void __launch_bounds__(256)
proj_qkv_kernel(const float* __restrict__ E,
                const float* __restrict__ WQ,
                const float* __restrict__ WK,
                const float* __restrict__ WV,
                unsigned short* __restrict__ Qb,
                unsigned short* __restrict__ Kb,
                unsigned short* __restrict__ VTb) {
  __shared__ __align__(16) unsigned short ldsA[64 * 32];
  __shared__ __align__(16) unsigned short ldsB[64 * 32];

  const int row0 = blockIdx.x * 64;
  const int col0 = blockIdx.y * 64;
  const int z    = blockIdx.z;
  const float* W = (z == 0) ? WQ : (z == 1) ? WK : WV;

  const int tid  = threadIdx.x;
  const int wave = tid >> 5;
  const int lane = tid & 31;
  const int lh   = lane >> 4;      // lane half
  const int ln   = lane & 15;
  const int mOff = (wave & 3) * 16;
  const int nOff = (wave >> 2) * 32;

  v8f acc0 = {}; v8f acc1 = {};

  for (int k0 = 0; k0 < DIM; k0 += 32) {
    #pragma unroll
    for (int j = 0; j < 8; ++j) {
      int i = tid + j * 256;               // 0..2047
      int r = i >> 5, k = i & 31;
      ldsA[i] = f2bf(E[(size_t)(row0 + r) * DIM + k0 + k]);
      ldsB[i] = f2bf(W[(size_t)(col0 + r) * DIM + k0 + k]);
    }
    __syncthreads();

    // A fragment (16x32 bf16): lane ln = row, 16 contiguous K per lane
    Frag a, b0, b1;
    #pragma unroll
    for (int e2 = 0; e2 < 8; ++e2) {
      int kk = 2 * e2 + lh * 16;
      a.u[e2] = *(const unsigned int*)&ldsA[(mOff + ln) * 32 + kk];
    }
    // B fragments (32x16): lane ln = column, 16 contiguous K per lane
    {
      const unsigned short* pb = &ldsB[(nOff + ln) * 32 + lh * 16];
      b0.q2[0] = *(const uint4*)(pb);
      b0.q2[1] = *(const uint4*)(pb + 8);
      pb = &ldsB[(nOff + 16 + ln) * 32 + lh * 16];
      b1.q2[0] = *(const uint4*)(pb);
      b1.q2[1] = *(const uint4*)(pb + 8);
    }
    acc0 = wmma_bf16(a, b0, acc0);
    acc1 = wmma_bf16(a, b1, acc1);
    __syncthreads();
  }

  // store: C layout lane->N=ln, vgpr v -> M = v + lh*8
  if (z < 2) {
    unsigned short* dst = (z == 0) ? Qb : Kb;
    #pragma unroll
    for (int v = 0; v < 8; ++v) {
      size_t r = (size_t)(row0 + mOff + v + lh * 8) * DIM;
      dst[r + col0 + nOff + ln]      = f2bf(acc0[v]);
      dst[r + col0 + nOff + 16 + ln] = f2bf(acc1[v]);
    }
  } else {
    // VT[b][d][s]: s contiguous across v -> pack 8 bf16 -> one b128 store
    int sg = row0 + mOff + lh * 8;
    int b  = sg >> 11;
    int s  = sg & (SEQ - 1);
    uint4 w0, w1;
    w0.x = f2bf(acc0[0]) | ((unsigned)f2bf(acc0[1]) << 16);
    w0.y = f2bf(acc0[2]) | ((unsigned)f2bf(acc0[3]) << 16);
    w0.z = f2bf(acc0[4]) | ((unsigned)f2bf(acc0[5]) << 16);
    w0.w = f2bf(acc0[6]) | ((unsigned)f2bf(acc0[7]) << 16);
    w1.x = f2bf(acc1[0]) | ((unsigned)f2bf(acc1[1]) << 16);
    w1.y = f2bf(acc1[2]) | ((unsigned)f2bf(acc1[3]) << 16);
    w1.z = f2bf(acc1[4]) | ((unsigned)f2bf(acc1[5]) << 16);
    w1.w = f2bf(acc1[6]) | ((unsigned)f2bf(acc1[7]) << 16);
    size_t base = (size_t)b * DIM * SEQ;
    *(uint4*)&VTb[base + (size_t)(col0 + nOff + ln) * SEQ + s]      = w0;
    *(uint4*)&VTb[base + (size_t)(col0 + nOff + 16 + ln) * SEQ + s] = w1;
  }
}

// ---------------------------------------------------------------------------
// Stage a 32x256 bf16 K tile (contiguous 16KB) into LDS.
// 1024 x b128 chunks over 128 threads -> 8 per thread. Async when available.
// ---------------------------------------------------------------------------
__device__ __forceinline__ void stage_k(const unsigned short* __restrict__ K,
                                        int kn, unsigned short* dst, int tid) {
  const unsigned short* src = K + (size_t)kn * DIM;
  #pragma unroll
  for (int j = 0; j < 8; ++j) {
    int c = tid + j * 128;                 // chunk id, 16B each
#if HAVE_ASYNC_LDS
    // AS1/AS3-qualified v4i pointers; materialize via uintptr_t round-trip
    // (low 32 bits of a generic LDS pointer are the LDS offset on CDNA5).
    __builtin_amdgcn_global_load_async_to_lds_b128(
        (gas_v4i)(uintptr_t)(src + c * 8),
        (las_v4i)(uintptr_t)(dst + c * 8), 0, 0);
#else
    *(uint4*)(dst + c * 8) = *(const uint4*)(src + c * 8);
#endif
  }
}

// ---------------------------------------------------------------------------
// Kernel 2: causal flash attention. grid = (S/64, B), block = 128 (4 waves).
// Each wave owns 16 query rows; block stages shared K tiles (double-buffered
// async copies), loops over 32-key steps up to the diagonal. Online softmax
// in fp32; P goes C-layout -> LDS -> A-layout; P x V against VT B-frags.
// ---------------------------------------------------------------------------
__global__ void __launch_bounds__(128)
flash_attn_kernel(const unsigned short* __restrict__ Qb,
                  const unsigned short* __restrict__ Kb,
                  const unsigned short* __restrict__ VTb,
                  float* __restrict__ out) {
  __shared__ __align__(16) unsigned short ldsK[2][32 * DIM];   // 2 x 16KB
  __shared__ __align__(16) unsigned short ldsP[4][16 * 32];    // 4KB

  const int b    = blockIdx.y;
  const int tid  = threadIdx.x;
  const int wave = tid >> 5;
  const int lane = tid & 31;
  const int lh   = lane >> 4;
  const int ln   = lane & 15;
  const int q0b  = blockIdx.x * 64;
  const int q0   = q0b + wave * 16;        // wave's first query row

  const unsigned short* Q  = Qb  + ((size_t)b * SEQ + q0) * DIM;
  const unsigned short* K  = Kb  + (size_t)b * SEQ * DIM;
  const unsigned short* VT = VTb + (size_t)b * DIM * SEQ;

  // Resident Q A-fragments: 8 k-steps of 16x32 (2 contiguous b128 each)
  Frag qf[8];
  #pragma unroll
  for (int ks = 0; ks < 8; ++ks) {
    const unsigned short* p = Q + (size_t)ln * DIM + ks * 32 + lh * 16;
    qf[ks].q2[0] = *(const uint4*)(p);        // e 0..7  -> d = base+lh*16+0..7
    qf[ks].q2[1] = *(const uint4*)(p + 8);    // e 8..15 -> d = base+lh*16+8..15
  }

  v8f acc[16];
  #pragma unroll
  for (int t = 0; t < 16; ++t) acc[t] = (v8f){};
  float mrow[8], lrow[8];
  #pragma unroll
  for (int v = 0; v < 8; ++v) { mrow[v] = -3.0e38f; lrow[v] = 0.0f; }

  const float scale = 0.0625f;   // 1/sqrt(256)
  unsigned short* P = ldsP[wave];
  const int kmaxb = q0b + 63;    // block-uniform causal bound

  stage_k(K, 0, ldsK[0], tid);   // prologue fill of buffer 0
  wait_async_all();
  __syncthreads();
  int buf = 0;

  for (int kn = 0; kn <= kmaxb; kn += 32) {
    // issue async copy of next tile into the other buffer (overlaps compute)
    if (kn + 32 <= kmaxb) stage_k(K, kn + 32, ldsK[buf ^ 1], tid);

    if (kn <= q0 + 15) {         // this wave still has causal work
      const unsigned short* Kt = ldsK[buf];
      // prefetch the next VT tile region (hint only)
      __builtin_prefetch(VT + (size_t)(lane * 8) * SEQ + kn + 32, 0, 1);

      // ---- scores: two 16x16 tiles (keys kn..+15, kn+16..+31) from LDS ----
      v8f c0 = {}, c1 = {};
      #pragma unroll
      for (int ks = 0; ks < 8; ++ks) {
        Frag b0, b1;
        const unsigned short* p0 = Kt + (size_t)ln * DIM + ks * 32 + lh * 16;
        const unsigned short* p1 = Kt + (size_t)(16 + ln) * DIM + ks * 32 + lh * 16;
        b0.q2[0] = *(const uint4*)(p0);
        b0.q2[1] = *(const uint4*)(p0 + 8);
        b1.q2[0] = *(const uint4*)(p1);
        b1.q2[1] = *(const uint4*)(p1 + 8);
        c0 = wmma_bf16(qf[ks], b0, c0);
        c1 = wmma_bf16(qf[ks], b1, c1);
      }

      // ---- causal mask (diagonal tiles only) + online softmax (fp32) ----
      const bool needMask = (kn + 31 > q0);
      float p0v[8], p1v[8];
      #pragma unroll
      for (int v = 0; v < 8; ++v) {
        int row = q0 + v + lh * 8;
        float s0 = c0[v] * scale;
        float s1 = c1[v] * scale;
        if (needMask) {
          if (kn + ln > row)      s0 = -3.0e38f;
          if (kn + 16 + ln > row) s1 = -3.0e38f;
        }
        float t = fmaxf(s0, s1);
        t = fmaxf(t, __shfl_xor(t, 1, 32));
        t = fmaxf(t, __shfl_xor(t, 2, 32));
        t = fmaxf(t, __shfl_xor(t, 4, 32));
        t = fmaxf(t, __shfl_xor(t, 8, 32));
        float mnew = fmaxf(mrow[v], t);
        float corr = __expf(mrow[v] - mnew);
        mrow[v] = mnew;
        float e0 = __expf(s0 - mnew);
        float e1 = __expf(s1 - mnew);
        float rs = e0 + e1;
        rs += __shfl_xor(rs, 1, 32);
        rs += __shfl_xor(rs, 2, 32);
        rs += __shfl_xor(rs, 4, 32);
        rs += __shfl_xor(rs, 8, 32);
        lrow[v] = lrow[v] * corr + rs;
        #pragma unroll
        for (int t2 = 0; t2 < 16; ++t2) acc[t2][v] *= corr;
        p0v[v] = e0;
        p1v[v] = e1;
      }

      // ---- P: C-layout -> private LDS scratch -> A-layout fragment ----
      #pragma unroll
      for (int v = 0; v < 8; ++v) {
        int m = v + lh * 8;
        P[m * 32 + ln]      = f2bf(p0v[v]);
        P[m * 32 + 16 + ln] = f2bf(p1v[v]);
      }
      asm volatile("s_wait_dscnt 0x0" ::: "memory");
      Frag pa;
      #pragma unroll
      for (int e2 = 0; e2 < 8; ++e2) {
        int kk = 2 * e2 + lh * 16;
        pa.u[e2] = *(const unsigned int*)&P[ln * 32 + kk];
      }

      // ---- P x V: 16 output d-tiles, VT gives contiguous-in-s B-frags ----
      #pragma unroll
      for (int t2 = 0; t2 < 16; ++t2) {
        const unsigned short* pv = VT + (size_t)(t2 * 16 + ln) * SEQ + kn + lh * 16;
        Frag bv;
        bv.q2[0] = *(const uint4*)(pv);
        bv.q2[1] = *(const uint4*)(pv + 8);
        acc[t2] = wmma_bf16(pa, bv, acc[t2]);
      }
    }

    // next tile's async copy must land; all waves done reading current buf
    wait_async_all();
    __syncthreads();
    buf ^= 1;
  }

  // ---- epilogue: out = acc / (l + EPS) ----
  float* O = out + ((size_t)b * SEQ + q0) * DIM;
  #pragma unroll
  for (int v = 0; v < 8; ++v) {
    float inv = 1.0f / (lrow[v] + 1e-9f);
    int m = v + lh * 8;
    #pragma unroll
    for (int t2 = 0; t2 < 16; ++t2)
      O[(size_t)m * DIM + t2 * 16 + ln] = acc[t2][v] * inv;
  }
}

// ---------------------------------------------------------------------------
extern "C" void kernel_launch(void* const* d_in, const int* in_sizes, int n_in,
                              void* d_out, int out_size, void* d_ws, size_t ws_size,
                              hipStream_t stream) {
  const float* E  = (const float*)d_in[0];
  const float* WQ = (const float*)d_in[1];
  const float* WK = (const float*)d_in[2];
  const float* WV = (const float*)d_in[3];

  unsigned short* Qb  = (unsigned short*)d_ws;                 // [16384][256] bf16
  unsigned short* Kb  = Qb + (size_t)NROWS * DIM;              // [16384][256] bf16
  unsigned short* VTb = Kb + (size_t)NROWS * DIM;              // [8][256][2048] bf16

  dim3 g1(NROWS / 64, DIM / 64, 3);
  proj_qkv_kernel<<<g1, 256, 0, stream>>>(E, WQ, WK, WV, Qb, Kb, VTb);

  dim3 g2(SEQ / 64, 8);
  flash_attn_kernel<<<g2, 128, 0, stream>>>(Qb, Kb, VTb, (float*)d_out);
}
